// SpinConvSq2d_12369505813005
// MI455X (gfx1250) — matrix operand order repaired and checked by reading, not compiled, not run on hardware
//
#include <hip/hip_runtime.h>

typedef __attribute__((ext_vector_type(16))) __bf16 v16bf;
typedef __attribute__((ext_vector_type(8)))  float  v8f;

// Constants (folded into weights by prep kernel)
#define SH0c   0.28209479177387814f   // 1/(2 sqrt(pi))
#define SH1c   0.4886025119029199f    // sqrt(3/(4 pi))
#define CG110c 0.5773502691896258f    // 1/sqrt(3)
#define CG111c 0.7071067811865476f    // 1/sqrt(2)
#define A0c    0.035355339059327376f  // 1/sqrt(800)
#define A1c    0.028867513459481287f  // 1/sqrt(1200)

union U16 { uint4 u[2]; v16bf v; };

// Dynamic LDS partition (bytes)
#define AS_BYTES   (5 * 64 * 72 * 2)            // 46080: feat rows, channel-permuted bf16
#define BS_BYTES   (160 * 56 * 2)               // 17920: one K-slab of weights
#define OFF_BS0    (AS_BYTES)                   // 46080
#define OFF_BS1    (OFF_BS0 + BS_BYTES)         // 64000
#define OFF_SS     (OFF_BS1 + BS_BYTES)         // 81920
#define SMEM_BYTES (OFF_SS + 64 * 3 * 4)        // 82688

#if defined(__AMDGCN__) && __has_builtin(__builtin_amdgcn_global_load_async_to_lds_b128) && \
    __has_builtin(__builtin_amdgcn_s_wait_asynccnt)
#define USE_ASYNC_LDS 1
typedef int v4i __attribute__((ext_vector_type(4)));
typedef __attribute__((address_space(1))) v4i* g_v4i_ptr;   // "__device__" side
typedef __attribute__((address_space(3))) v4i* l_v4i_ptr;   // "__shared__" side
#else
#define USE_ASYNC_LDS 0
#endif

// ---------------------------------------------------------------------------
// Prep: folded bf16 weight matrix Bw[n][k], n=0..159, k=0..415 (k>=400 zero)
//   n  0..31 : A0*SH0   * W_000      n 32..63 : A1        * W_011
//   n 64..95 : A0*CG110 * W_110      n 96..127: A1*SH0    * W_101
//   n128..159: A1*CG111 * W_111        (k = a*16+u; W[a][u][w] = W[k*32+w])
// ---------------------------------------------------------------------------
__global__ __launch_bounds__(256) void prep_weights(
    const float* __restrict__ W000, const float* __restrict__ W110,
    const float* __restrict__ W011, const float* __restrict__ W101,
    const float* __restrict__ W111, __bf16* __restrict__ Bw)
{
    const int idx = blockIdx.x * 256 + threadIdx.x;
    if (idx >= 160 * 416) return;
    const int nrow = idx / 416;
    const int k    = idx - nrow * 416;
    float v = 0.0f;
    if (k < 400) {
        const int w   = nrow & 31;
        const int blk = nrow >> 5;
        const float* W; float sc;
        switch (blk) {
            case 0:  W = W000; sc = A0c * SH0c;   break;
            case 1:  W = W011; sc = A1c;          break;
            case 2:  W = W110; sc = A0c * CG110c; break;
            case 3:  W = W101; sc = A1c * SH0c;   break;
            default: W = W111; sc = A1c * CG111c; break;
        }
        v = sc * W[k * 32 + w];
    }
    Bw[nrow * 416 + k] = (__bf16)v;
}

// ---------------------------------------------------------------------------
// Main: one 128-thread block per (n,x) row. Wave w owns sites y in [16w,16w+16).
// K=400 (25 taps x 16 u) x N=160 folded-weight GEMM in bf16 WMMA (fp32 acc),
// double-buffered async weight slabs, then per-site spin recombination.
// ---------------------------------------------------------------------------
__global__ __launch_bounds__(128) void spinconv_main(
    const float* __restrict__ feat, const float* __restrict__ spin,
    const __bf16* __restrict__ Bw, float* __restrict__ out)
{
    extern __shared__ char smem[];
    __bf16* As  = (__bf16*)smem;
    __bf16* Bs0 = (__bf16*)(smem + OFF_BS0);
    __bf16* Bs1 = (__bf16*)(smem + OFF_BS1);
    float*  Ss  = (float*)(smem + OFF_SS);

    const int tid  = threadIdx.x;
    const int lane = tid & 31;
    const int wave = tid >> 5;
    const int n    = blockIdx.x >> 6;
    const int x    = blockIdx.x & 63;

    // ---- stage As: feat[n, x-2..x+2, :, :] -> bf16, channels permuted to
    //      [x0(16) | y_x(16) | y_y(16) | y_z(16)] ----
    for (int idx = tid; idx < 5 * 64 * 64; idx += 128) {
        const int r = idx >> 12;
        const int y = (idx >> 6) & 63;
        const int c = idx & 63;
        const int xr = (x + r - 2) & 63;
        const float v = feat[((((n << 6) + xr) << 6) + y) * 64 + c];
        int cp;
        if (c < 16) cp = c;
        else { const int t = c - 16; const int u = t / 3; const int g = t - 3 * u;
               cp = 16 + (g << 4) + u; }
        As[(r * 64 + y) * 72 + cp] = (__bf16)v;
    }
    if (tid < 64) {
        const float* sp = spin + ((size_t)((n << 6) + x) * 64 + tid) * 3;
        Ss[tid * 3 + 0] = SH1c * sp[0];
        Ss[tid * 3 + 1] = SH1c * sp[1];
        Ss[tid * 3 + 2] = SH1c * sp[2];
    }

    const int m     = lane & 15;      // M row within 16x16 tile == site index
    const int half  = lane >> 4;      // K-half for A/B fragments
    const int ysite = (wave << 4) + m;
    const int coff  = half << 3;

    // 22 accumulators: [0..1]=g (W000), [2..3]=q (W011),
    // [4+t*3+i]=h_i, [10+t*3+i]=p_i, [16+t*3+i]=c_i   (t = column-subtile 0..1)
    v8f acc[22];
    const v8f vzero = {0.f,0.f,0.f,0.f,0.f,0.f,0.f,0.f};
#pragma unroll
    for (int t = 0; t < 22; ++t) acc[t] = vzero;

    // ---- per-step compute: A fragments + 22 WMMAs against slab Bsc ----
    auto compute_step = [&](int s, const __bf16* Bsc) {
        const int  t0   = 2 * s;
        const int  t1   = t0 + 1;
        const bool last = (s == 12);                 // t1 == 25 -> zero pad
        const int  r0   = t0 / 5;
        const int  yA   = (ysite + (t0 - 5 * r0) - 2) & 63;
        U16 fa[4];
#pragma unroll
        for (int g = 0; g < 4; ++g)
            fa[g].u[0] = *(const uint4*)(&As[(r0 * 64 + yA) * 72 + (g << 4) + coff]);
        if (!last) {
            const int r1 = t1 / 5;
            const int yB = (ysite + (t1 - 5 * r1) - 2) & 63;
#pragma unroll
            for (int g = 0; g < 4; ++g)
                fa[g].u[1] = *(const uint4*)(&As[(r1 * 64 + yB) * 72 + (g << 4) + coff]);
        } else {
#pragma unroll
            for (int g = 0; g < 4; ++g) fa[g].u[1] = make_uint4(0u, 0u, 0u, 0u);
        }
#pragma unroll
        for (int nt = 0; nt < 10; ++nt) {
            U16 fb;
            const __bf16* pb = &Bsc[(nt * 16 + m) * 56 + (half << 3)];
            fb.u[0] = *(const uint4*)(pb);
            fb.u[1] = *(const uint4*)(pb + 16);
            if (nt < 4) {
                acc[nt] = __builtin_amdgcn_wmma_f32_16x16x32_bf16(
                    false, fa[0].v, false, fb.v, (short)0, acc[nt], false, false);
            } else {
                const int b = 4 + (nt - 4) * 3;
                acc[b + 0] = __builtin_amdgcn_wmma_f32_16x16x32_bf16(
                    false, fa[1].v, false, fb.v, (short)0, acc[b + 0], false, false);
                acc[b + 1] = __builtin_amdgcn_wmma_f32_16x16x32_bf16(
                    false, fa[2].v, false, fb.v, (short)0, acc[b + 1], false, false);
                acc[b + 2] = __builtin_amdgcn_wmma_f32_16x16x32_bf16(
                    false, fa[3].v, false, fb.v, (short)0, acc[b + 2], false, false);
            }
        }
    };

#if USE_ASYNC_LDS
    // ---- double-buffered weight slabs via async global->LDS DMA ----
    auto issue_slab = [&](int s, __bf16* dst) {
        const int k0 = s << 5;
#pragma unroll
        for (int j = 0; j < 5; ++j) {
            const int idx = tid + (j << 7);          // 640 = 160 rows * 4 chunks
            const int row = idx >> 2;
            const int q   = idx & 3;
            void* gv = (void*)(Bw + row * 416 + k0 + (q << 3));
            void* lv = (void*)(dst + row * 56 + (q << 3));
            __builtin_amdgcn_global_load_async_to_lds_b128(
                (g_v4i_ptr)gv, (l_v4i_ptr)lv, 0, 0);
        }
    };
    issue_slab(0, Bs0);
    __builtin_amdgcn_s_wait_asynccnt(0);
    __syncthreads();
    for (int s = 0; s < 13; ++s) {
        if (s < 12) issue_slab(s + 1, (s & 1) ? Bs0 : Bs1);
        compute_step(s, (s & 1) ? Bs1 : Bs0);
        if (s < 12) {
            __builtin_amdgcn_s_wait_asynccnt(0);
            __syncthreads();
        }
    }
#else
    // ---- fallback: register-pipelined double buffer ----
    uint4 rb[5];
    auto fetch_slab = [&](int s) {
        const int k0 = s << 5;
#pragma unroll
        for (int j = 0; j < 5; ++j) {
            const int idx = tid + (j << 7);
            const int row = idx >> 2;
            const int q   = idx & 3;
            rb[j] = *(const uint4*)(Bw + row * 416 + k0 + (q << 3));
        }
    };
    auto store_slab = [&](__bf16* dst) {
#pragma unroll
        for (int j = 0; j < 5; ++j) {
            const int idx = tid + (j << 7);
            const int row = idx >> 2;
            const int q   = idx & 3;
            *(uint4*)(&dst[row * 56 + (q << 3)]) = rb[j];
        }
    };
    fetch_slab(0);
    store_slab(Bs0);
    __syncthreads();
    for (int s = 0; s < 13; ++s) {
        if (s < 12) fetch_slab(s + 1);
        compute_step(s, (s & 1) ? Bs1 : Bs0);
        if (s < 12) {
            store_slab((s & 1) ? Bs0 : Bs1);
            __syncthreads();
        }
    }
#endif

    // ---- per-site spin recombination + store (C/D layout: lane = column,
    //      element e = M row) ----
    const int col   = m;
    const int rbase = half << 3;
    const int ybase = wave << 4;
    float* orow = out + (size_t)((n << 6) + x) * 64 * 128;
#pragma unroll
    for (int e = 0; e < 8; ++e) {
        const int yy = ybase + rbase + e;
        const float sx = Ss[yy * 3 + 0];
        const float sy = Ss[yy * 3 + 1];
        const float sz = Ss[yy * 3 + 2];
        float* o = orow + (size_t)yy * 128;
#pragma unroll
        for (int t = 0; t < 2; ++t) {
            const int w = (t << 4) + col;
            const float hx = acc[4 + t * 3 + 0][e];
            const float hy = acc[4 + t * 3 + 1][e];
            const float hz = acc[4 + t * 3 + 2][e];
            o[w] = acc[t][e] + sx * hx + sy * hy + sz * hz;
            const float q  = acc[2 + t][e];
            const float px = acc[10 + t * 3 + 0][e];
            const float py = acc[10 + t * 3 + 1][e];
            const float pz = acc[10 + t * 3 + 2][e];
            const float cx = acc[16 + t * 3 + 0][e];
            const float cy = acc[16 + t * 3 + 1][e];
            const float cz = acc[16 + t * 3 + 2][e];
            o[32 + w * 3 + 0] = q * sx + px + (sz * cy - sy * cz);
            o[32 + w * 3 + 1] = q * sy + py + (sx * cz - sz * cx);
            o[32 + w * 3 + 2] = q * sz + pz + (sy * cx - sx * cy);
        }
    }
}

// ---------------------------------------------------------------------------
extern "C" void kernel_launch(void* const* d_in, const int* in_sizes, int n_in,
                              void* d_out, int out_size, void* d_ws, size_t ws_size,
                              hipStream_t stream)
{
    const float* feat = (const float*)d_in[0];
    const float* spin = (const float*)d_in[1];
    const float* W000 = (const float*)d_in[2];
    const float* W110 = (const float*)d_in[3];
    const float* W011 = (const float*)d_in[4];
    const float* W101 = (const float*)d_in[5];
    const float* W111 = (const float*)d_in[6];
    __bf16* Bw = (__bf16*)d_ws;          // needs 160*416*2 = 133,120 bytes
    float*  out = (float*)d_out;

    // Allow >64KB dynamic LDS (WGP has 320KB); non-stream API, graph-capture safe.
    (void)hipFuncSetAttribute((const void*)spinconv_main,
                              hipFuncAttributeMaxDynamicSharedMemorySize, SMEM_BYTES);

    prep_weights<<<(160 * 416 + 255) / 256, 256, 0, stream>>>(W000, W110, W011, W101, W111, Bw);
    spinconv_main<<<8 * 64, 128, SMEM_BYTES, stream>>>(feat, spin, Bw, out);
}
